// BasicLSTM_58926951301291
// MI455X (gfx1250) — matrix operand order repaired and checked by reading, not compiled, not run on hardware
//
#include <hip/hip_runtime.h>
#include <hip/hip_bf16.h>

// B=512, T=1024, D=128, H=256, gates 4H=1024
// Layer1 GEMM: [16 x 384] * [384 x 1024]  (K tiles: 12)
// Layer2 GEMM: [16 x 512] * [512 x 1024]  (K tiles: 16)

typedef __attribute__((ext_vector_type(16))) __bf16 v16bf;
typedef __attribute__((ext_vector_type(8)))  float  v8f;

#define T_STEPS 1024
#define KC1 12              // 384 / 32
#define KC2 16              // 512 / 32
#define NT_TOTAL 64         // 1024 / 16
#define W1_HALFS (KC1 * NT_TOTAL * 512)   // 393216
#define W2_HALFS (KC2 * NT_TOTAL * 512)   // 524288

// ---- helpers -------------------------------------------------------------

// A-matrix 16x32 bf16 VGPR layout (same K order as B):
// lanes 0-15 hold K=0..15 across V0..7 (pairs); lanes16-31 hold K=16..31
__device__ __forceinline__ int k_of_a(int lane, int e) {
  int v = e >> 1, lo = e & 1;
  return 2 * v + lo + ((lane & 16) ? 16 : 0);
}

// B-matrix 32x16 bf16 VGPR layout used here (contiguous K, same order as A):
// lanes 0-15 hold K=0..15 across V0..7 (pairs); lanes16-31 hold K=16..31
__device__ __forceinline__ int k_of_b(int lane, int e) {
  int v = e >> 1, lo = e & 1;
  return 2 * v + lo + ((lane & 16) ? 16 : 0);
}

__device__ __forceinline__ float sigm(float v) {
  return 1.0f / (1.0f + __expf(-v));
}
__device__ __forceinline__ float tanh_fast(float v) {
  v = fminf(fmaxf(v, -15.0f), 15.0f);
  float e = __expf(2.0f * v);
  return (e - 1.0f) / (e + 1.0f);
}

// ---- prep: fp32 weights -> bf16 tiles, layout [kchunk][ntile][lane][e] ----
__global__ void lstm_prep_kernel(const float* __restrict__ W1,
                                 const float* __restrict__ W2,
                                 __bf16* __restrict__ ws) {
  int idx = blockIdx.x * 256 + threadIdx.x;
  if (idx < W1_HALFS) {
    int e = idx & 15, lane = (idx >> 4) & 31;
    int nt = (idx >> 9) & 63, kc = idx >> 15;       // 64*512 = 32768 per kc
    int k = kc * 32 + k_of_b(lane, e);
    int n = nt * 16 + (lane & 15);
    ws[idx] = (__bf16)W1[k * 1024 + n];             // W1 is [384 x 1024]
  } else if (idx < W1_HALFS + W2_HALFS) {
    int r = idx - W1_HALFS;
    int e = r & 15, lane = (r >> 4) & 31;
    int nt = (r >> 9) & 63, kc = r >> 15;
    int k = kc * 32 + k_of_b(lane, e);
    int n = nt * 16 + (lane & 15);
    ws[idx] = (__bf16)W2[k * 1024 + n];             // W2 is [512 x 1024]
  }
}

// ---- main persistent recurrence kernel -----------------------------------
// grid = 32 blocks (one per 16-row batch tile), block = 256 threads = 8 waves.
// Wave w owns gate-column tiles {g*16 + 2w, g*16 + 2w + 1 : g in 0..3} so the
// f/i/g/o partials for its j-columns live in its own 8 accumulators.
__global__ void __launch_bounds__(256, 1)
lstm_main_kernel(const float* __restrict__ x,
                 const float* __restrict__ b1,
                 const float* __restrict__ b2,
                 const float* __restrict__ Wout,
                 const float* __restrict__ bout,
                 const __bf16* __restrict__ wbf,
                 float* __restrict__ out) {
  __shared__ __align__(32) __bf16 A_lds[KC2 * 512];  // 16 KB (max K tiles)
  __shared__ float h1s[16 * 256];                    // 16 KB
  __shared__ float h2s[16 * 256];                    // 16 KB

  const int tid  = threadIdx.x;
  const int lane = tid & 31;
  const int wave = tid >> 5;
  const int row0 = blockIdx.x * 16;

  const v16bf* W1t = reinterpret_cast<const v16bf*>(wbf);             // [12][64][32]
  const v16bf* W2t = reinterpret_cast<const v16bf*>(wbf + W1_HALFS);  // [16][64][32]
  const v16bf* A_v = reinterpret_cast<const v16bf*>(A_lds);           // [kc][32]

  float c1r[16], c2r[16];
#pragma unroll
  for (int i = 0; i < 16; ++i) { c1r[i] = 0.0f; c2r[i] = 0.0f; }
  for (int i = tid; i < 16 * 256; i += 256) { h1s[i] = 0.0f; h2s[i] = 0.0f; }

  // per-wave bias registers: acc i -> gate g=i>>1, j-tile p=i&1
  float bias1[8], bias2[8];
#pragma unroll
  for (int i = 0; i < 8; ++i) {
    int n = (i >> 1) * 256 + (2 * wave + (i & 1)) * 16 + (lane & 15);
    bias1[i] = b1[n];
    bias2[i] = b2[n];
  }

  // ---- precompute t-invariant A-build source offsets (register arrays) ----
  // s1off[it] >= 0 : x element at  x[s1off + t*128]
  // s1off[it] <  0 : h1 element at h1s[-1 - s1off]
  int s1off[24];
#pragma unroll
  for (int it = 0; it < 24; ++it) {
    int i = tid + it * 256;                 // 12*512 = 6144 halfs
    int e = i & 15, ln = (i >> 4) & 31, kc = i >> 9;
    int k = kc * 32 + k_of_a(ln, e);
    int m = ln & 15;
    s1off[it] = (k < 128) ? ((row0 + m) * (T_STEPS * 128) + k)
                          : (-1 - (m * 256 + (k - 128)));
  }
  // s2off[it] >= 0 : h1s[s2off] ; < 0 : h2s[-1 - s2off]
  int s2off[32];
#pragma unroll
  for (int it = 0; it < 32; ++it) {
    int i = tid + it * 256;                 // 16*512 = 8192 halfs
    int e = i & 15, ln = (i >> 4) & 31, kc = i >> 9;
    int k = kc * 32 + k_of_a(ln, e);
    int m = ln & 15;
    s2off[it] = (k < 256) ? (m * 256 + k) : (-1 - (m * 256 + (k - 256)));
  }
  __syncthreads();

  for (int t = 0; t < T_STEPS; ++t) {
    // Block loop-invariant load motion across timesteps: weights must be
    // re-streamed from (L2-hot) global each step, never preloaded+spilled.
    asm volatile("" ::: "memory");

    // ---- build A1 = concat(x_t, h1) in WMMA-A layout (12 K-chunks) ----
    const float* xt = x + (size_t)t * 128;
#pragma unroll
    for (int it = 0; it < 24; ++it) {
      int o = s1off[it];
      float v;
      if (o >= 0) {
        v = xt[o];
        if (t + 1 < T_STEPS) __builtin_prefetch(&xt[o + 128], 0, 1);
      } else {
        v = h1s[-1 - o];
      }
      A_lds[tid + it * 256] = (__bf16)v;
    }
    __syncthreads();

    // ---- GEMM1 (K=384) + cell1, all in registers ----
    {
      v8f acc[8] = {};
      const v16bf* bptr = W1t + (2 * wave) * 32 + lane;
#pragma unroll 2
      for (int kc = 0; kc < KC1; ++kc) {
        // batch-issue all 9 operand loads (1 A + 8 B) -> one wait, 8 WMMAs
        v16bf a = A_v[kc * 32 + lane];
        v16bf bt[8];
#pragma unroll
        for (int i = 0; i < 8; ++i)
          bt[i] = bptr[((i >> 1) * 16 + (i & 1)) * 32];   // const ioffsets
#pragma unroll
        for (int i = 0; i < 8; ++i)
          acc[i] = __builtin_amdgcn_wmma_f32_16x16x32_bf16(
              false, a, false, bt[i], (short)0, acc[i], false, false);
        bptr += NT_TOTAL * 32;              // advance one K-chunk
      }
#pragma unroll
      for (int p = 0; p < 2; ++p) {
        int j = (2 * wave + p) * 16 + (lane & 15);
#pragma unroll
        for (int r = 0; r < 8; ++r) {
          float gf = acc[0 + p][r] + bias1[0 + p];
          float gi = acc[2 + p][r] + bias1[2 + p];
          float gg = acc[4 + p][r] + bias1[4 + p];
          float go = acc[6 + p][r] + bias1[6 + p];
          float f = sigm(gf), ii = sigm(gi), g = tanh_fast(gg), o = sigm(go);
          int ci = p * 8 + r;
          float nc = f * c1r[ci] + ii * g;
          c1r[ci] = nc;
          int m = r + ((lane >> 4) << 3);   // C/D layout: M = r + 8*(lane/16)
          h1s[m * 256 + j] = o * tanh_fast(nc);
        }
      }
    }
    __syncthreads();

    // ---- build A2 = concat(h1_new, h2_old) (16 K-chunks) ----
#pragma unroll
    for (int it = 0; it < 32; ++it) {
      int o = s2off[it];
      float v = (o >= 0) ? h1s[o] : h2s[-1 - o];
      A_lds[tid + it * 256] = (__bf16)v;
    }
    __syncthreads();

    // ---- GEMM2 (K=512) + cell2 ----
    {
      v8f acc[8] = {};
      const v16bf* bptr = W2t + (2 * wave) * 32 + lane;
#pragma unroll 2
      for (int kc = 0; kc < KC2; ++kc) {
        v16bf a = A_v[kc * 32 + lane];
        v16bf bt[8];
#pragma unroll
        for (int i = 0; i < 8; ++i)
          bt[i] = bptr[((i >> 1) * 16 + (i & 1)) * 32];
#pragma unroll
        for (int i = 0; i < 8; ++i)
          acc[i] = __builtin_amdgcn_wmma_f32_16x16x32_bf16(
              false, a, false, bt[i], (short)0, acc[i], false, false);
        bptr += NT_TOTAL * 32;
      }
#pragma unroll
      for (int p = 0; p < 2; ++p) {
        int j = (2 * wave + p) * 16 + (lane & 15);
#pragma unroll
        for (int r = 0; r < 8; ++r) {
          float gf = acc[0 + p][r] + bias2[0 + p];
          float gi = acc[2 + p][r] + bias2[2 + p];
          float gg = acc[4 + p][r] + bias2[4 + p];
          float go = acc[6 + p][r] + bias2[6 + p];
          float f = sigm(gf), ii = sigm(gi), g = tanh_fast(gg), o = sigm(go);
          int ci = p * 8 + r;
          float nc = f * c2r[ci] + ii * g;
          c2r[ci] = nc;
          int m = r + ((lane >> 4) << 3);
          h2s[m * 256 + j] = o * tanh_fast(nc);
        }
      }
    }
    __syncthreads();
  }

  // ---- output: h2_T @ Wout + bout  ([16 x 256] * [256 x 1]) ----
  if (tid < 16) {
    float s = 0.0f;
    for (int j = 0; j < 256; ++j) s += h2s[tid * 256 + j] * Wout[j];
    out[row0 + tid] = s + bout[0];
  }
}

// ---- host glue -----------------------------------------------------------
extern "C" void kernel_launch(void* const* d_in, const int* in_sizes, int n_in,
                              void* d_out, int out_size, void* d_ws, size_t ws_size,
                              hipStream_t stream) {
  (void)in_sizes; (void)n_in; (void)out_size; (void)ws_size;
  const float* x    = (const float*)d_in[0];
  const float* W1   = (const float*)d_in[1];
  const float* b1   = (const float*)d_in[2];
  const float* W2   = (const float*)d_in[3];
  const float* b2   = (const float*)d_in[4];
  const float* Wout = (const float*)d_in[5];
  const float* bout = (const float*)d_in[6];
  float* out = (float*)d_out;
  __bf16* wbf = (__bf16*)d_ws;                   // needs 1,835,008 bytes

  // 1) swizzle weights to bf16 WMMA tiles (L2-resident afterwards)
  int total = W1_HALFS + W2_HALFS;               // 917504
  lstm_prep_kernel<<<(total + 255) / 256, 256, 0, stream>>>(W1, W2, wbf);

  // 2) persistent recurrence: 32 blocks x 256 threads (8 wave32 waves)
  lstm_main_kernel<<<32, 256, 0, stream>>>(x, b1, b2, Wout, bout, wbf, out);
}
